// TopoPool_13460427505958
// MI455X (gfx1250) — compile-verified, matrix operand order checked by reading, not run on hardware
//
#include <hip/hip_runtime.h>
#include <stdint.h>

#define NN 65536      // nodes
#define NE 1048576    // edges
#define CH 256        // in channels
#define NG 256        // graphs
#define WSB 120       // persistent watershed blocks (co-resident on MI455X)
#define SORT_TILE 4096

typedef __attribute__((ext_vector_type(16))) _Float16 v16h;
typedef __attribute__((ext_vector_type(8)))  float    v8f;
typedef unsigned __attribute__((ext_vector_type(4)))  v4u;
typedef int      __attribute__((ext_vector_type(4)))  v4i;
typedef int      __attribute__((ext_vector_type(8)))  v8i;

#if defined(__gfx1250__) && __has_builtin(__builtin_amdgcn_tensor_load_to_lds) && \
    __has_builtin(__builtin_amdgcn_s_wait_tensorcnt)
#define USE_TDM 1
#else
#define USE_TDM 0
#endif

// ---------- order-preserving float<->uint encoding for atomicMax ----------
__device__ __forceinline__ unsigned fenc(float x) {
  unsigned u = __float_as_uint(x);
  return (u & 0x80000000u) ? ~u : (u | 0x80000000u);
}
__device__ __forceinline__ float fdec(unsigned u) {
  unsigned v = (u & 0x80000000u) ? (u & 0x7fffffffu) : ~u;
  return __uint_as_float(v);
}

// ---------- init ----------
__global__ void init_small_kernel(int* S) {
  // zero scalars + pk + gstart + present + rank + tg + mg + sg (2304 ints)
  for (int i = threadIdx.x; i < 2304; i += 256) S[i] = 0;
  __syncthreads();
  if (threadIdx.x == 0) S[4] = NN;  // remaining unvisited nodes
}

__global__ void init_nc_kernel(unsigned* pooled_enc) {
  int64_t tot = (int64_t)NN * CH;
  int64_t stride = (int64_t)gridDim.x * blockDim.x;
  for (int64_t i = (int64_t)blockIdx.x * blockDim.x + threadIdx.x; i < tot; i += stride)
    pooled_enc[i] = 0u;
}

__global__ void init_n_kernel(int* visited, int* cl, int* fr, int* nf, int* pool_id,
                              unsigned* promits, int* cb_i, float* epool) {
  int i = blockIdx.x * blockDim.x + threadIdx.x;
  if (i >= NN) return;
  visited[i] = 0; cl[i] = 0; fr[i] = 0; nf[i] = 0;
  pool_id[i] = 0; promits[i] = 0u; cb_i[i] = 0; epool[i] = 0.f;
}

// ---------- graph ranges (batch is sorted) ----------
__global__ void starts_kernel(const int* __restrict__ batch, int* __restrict__ gstart) {
  int g = blockIdx.x * blockDim.x + threadIdx.x;
  if (g > NG) return;
  if (g == NG) { gstart[g] = NN; return; }
  int lo = 0, hi = NN;
  while (lo < hi) { int mid = (lo + hi) >> 1; if (batch[mid] < g) lo = mid + 1; else hi = mid; }
  gstart[g] = lo;
}

// ---------- per-graph mean (block per graph, thread per channel) ----------
__global__ void mean_kernel(const float* __restrict__ x, const int* __restrict__ gstart,
                            float* __restrict__ meanb) {
  int g = blockIdx.x, c = threadIdx.x;
  int s = gstart[g], e = gstart[g + 1];
  float acc = 0.f;
  for (int r = s; r < e; ++r) {
    acc += x[(size_t)r * CH + c];
    if (r + 8 < e) __builtin_prefetch(&x[(size_t)(r + 8) * CH + c], 0, 0);  // global_prefetch_b8
  }
  int cnt = e - s;
  meanb[g * CH + c] = acc / (float)(cnt > 0 ? cnt : 1);
}

// ---------- t[g] = mean[g] . W[CH:2CH] + b ----------
__global__ void tgraph_kernel(const float* __restrict__ meanb, const float* __restrict__ W,
                              const float* __restrict__ bias, float* __restrict__ tg) {
  __shared__ float red[256];
  int g = blockIdx.x, c = threadIdx.x;
  red[c] = meanb[g * CH + c] * W[CH + c];
  __syncthreads();
  for (int off = 128; off; off >>= 1) { if (c < off) red[c] += red[c + off]; __syncthreads(); }
  if (c == 0) tg[g] = red[0] + bias[0];
}

// ---------- elevation GEMV via WMMA: e_raw[i] = x[i] . W[0:CH] + t[batch[i]] ----------
// A = 16 node rows x K=32 f16 (documented layout), B = w_lo chunk replicated across cols.
// TDM path: each wave DMAs its 16x64-f32 sub-tile into LDS (tensor_load_to_lds),
// waits on TENSORcnt, converts from LDS, and feeds v_wmma.
__global__ void elev_dot_wmma_kernel(const float* __restrict__ x, const float* __restrict__ W,
                                     const int* __restrict__ batch, const float* __restrict__ tg,
                                     float* __restrict__ e_raw) {
  const int wave = threadIdx.x >> 5;          // 8 waves per block
  const int lane = threadIdx.x & 31;
  const int tile = blockIdx.x * 8 + wave;     // 16-node tile
  const int row0 = tile * 16;
  const int grp  = lane >> 4;                 // lane group 0/1
  const int m    = lane & 15;                 // node row within tile

  v8f c = {};

#if USE_TDM
  __shared__ float sA[8][16 * 64];            // 4 KB per wave, 32 KB per block
  float* shw = &sA[wave][0];
  unsigned long long base_ga = (unsigned long long)(uintptr_t)(x + (size_t)row0 * CH);
  unsigned ldsa = (unsigned)(uintptr_t)shw;   // LDS byte address (low 32 bits of flat)

  for (int kbc = 0; kbc < CH; kbc += 64) {
    unsigned long long ga = base_ga + (unsigned long long)kbc * 4ull;
    // D# group 0: count=1 | lds_addr | global_addr | type=2
    unsigned g0w0 = (unsigned)__builtin_amdgcn_readfirstlane(1);
    unsigned g0w1 = (unsigned)__builtin_amdgcn_readfirstlane((int)ldsa);
    unsigned g0w2 = (unsigned)__builtin_amdgcn_readfirstlane((int)(unsigned)(ga & 0xffffffffull));
    unsigned g0w3 = (unsigned)__builtin_amdgcn_readfirstlane(
        (int)(((unsigned)(ga >> 32) & 0x01ffffffu) | (2u << 30)));
    v4u g0; g0[0] = g0w0; g0[1] = g0w1; g0[2] = g0w2; g0[3] = g0w3;

    // D# group 1: data_size=4B; tensor_dim0=256; tensor_dim1=65536; tile=64x16; stride0=256
    v8i g1;
    g1[0] = (int)(2u << 16);                                   // data_size = 4 bytes
    g1[1] = (int)(((unsigned)CH & 0xffffu) << 16);             // tensor_dim0 lo16
    g1[2] = (int)(((unsigned)CH >> 16) | (((unsigned)NN & 0xffffu) << 16));
    g1[3] = (int)(((unsigned)NN >> 16) | (64u << 16));         // tensor_dim1 hi16 | tile_dim0=64
    g1[4] = 16;                                                // tile_dim1=16, tile_dim2=0
    g1[5] = CH;                                                // tensor_dim0_stride lo32
    g1[6] = 0;
    g1[7] = 0;
    v4i gz = {0, 0, 0, 0};
#if defined(__clang_major__) && __clang_major__ >= 23
    v8i gz8 = {0, 0, 0, 0, 0, 0, 0, 0};
    __builtin_amdgcn_tensor_load_to_lds(g0, g1, gz, gz, gz8, 0);
#else
    __builtin_amdgcn_tensor_load_to_lds(g0, g1, gz, gz, 0);
#endif
    __builtin_amdgcn_s_wait_tensorcnt(0);

    for (int kb2 = 0; kb2 < 64; kb2 += 32) {
      v16h a, bm;
      const float* xr = &shw[m * 64 + kb2];
      const int kb = kbc + kb2;
#pragma unroll
      for (int h = 0; h < 8; ++h) {
        int k0 = (grp ? 8 : 0) + h;
        int k1 = (grp ? 24 : 16) + h;
        a[h]      = (_Float16)xr[k0];
        a[h + 8]  = (_Float16)xr[k1];
        bm[h]     = (_Float16)W[kb + k0];
        bm[h + 8] = (_Float16)W[kb + k1];
      }
      c = __builtin_amdgcn_wmma_f32_16x16x32_f16(false, a, false, bm, (short)0, c, false, false);
    }
  }
#else
  for (int kb = 0; kb < CH; kb += 32) {
    v16h a, bm;
    const float* xr = x + (size_t)(row0 + m) * CH + kb;
#pragma unroll
    for (int h = 0; h < 8; ++h) {
      int k0 = (grp ? 8 : 0) + h;
      int k1 = (grp ? 24 : 16) + h;
      a[h]      = (_Float16)xr[k0];
      a[h + 8]  = (_Float16)xr[k1];
      bm[h]     = (_Float16)W[kb + k0];
      bm[h + 8] = (_Float16)W[kb + k1];
    }
    c = __builtin_amdgcn_wmma_f32_16x16x32_f16(false, a, false, bm, (short)0, c, false, false);
  }
#endif

  // D layout: VGPR r -> M = r + 8*grp; column N = lane&15. Take column 0 (lanes 0 and 16).
  if ((lane & 15) == 0) {
    int mbase = row0 + grp * 8;
#pragma unroll
    for (int r = 0; r < 8; ++r) {
      int node = mbase + r;
      e_raw[node] = c[r] + tg[batch[node]];
    }
  }
}

// ---------- scatter softmax over batch (contiguous ranges), writes elev + r4 ----------
__global__ void softmax_batch_kernel(const float* __restrict__ e_raw, const int* __restrict__ gstart,
                                     float* __restrict__ out_elev, float* __restrict__ r4) {
  __shared__ float red[256];
  int g = blockIdx.x, t = threadIdx.x;
  int s = gstart[g], e = gstart[g + 1];
  float m = -INFINITY;
  for (int i = s + t; i < e; i += 256) m = fmaxf(m, e_raw[i]);
  red[t] = m; __syncthreads();
  for (int off = 128; off; off >>= 1) { if (t < off) red[t] = fmaxf(red[t], red[t + off]); __syncthreads(); }
  m = red[0];
  if (!(m * 0.0f == 0.0f)) m = 0.0f;  // isfinite guard, matches reference
  __syncthreads();
  float a = 0.f;
  for (int i = s + t; i < e; i += 256) a += expf(e_raw[i] - m);
  red[t] = a; __syncthreads();
  for (int off = 128; off; off >>= 1) { if (t < off) red[t] += red[t + off]; __syncthreads(); }
  float sden = red[0];
  sden = (sden > 0.f) ? sden : 1.f;
  for (int i = s + t; i < e; i += 256) {
    float ev = expf(e_raw[i] - m) / sden;
    out_elev[i] = ev;
    r4[i] = rintf(ev * 1e4f) * 1e-4f;   // jnp.round(.,4) ~ RNE
  }
}

// ---------- atomic grid barrier for persistent watershed kernel ----------
__device__ __forceinline__ void gsync(int* S, int nb) {
  __syncthreads();
  if (threadIdx.x == 0) {
    __threadfence();
    volatile int* vgen = S + 1;
    int g = *vgen;
    int a = atomicAdd(S + 0, 1);
    if (a == nb - 1) { S[0] = 0; __threadfence(); atomicAdd(S + 1, 1); }
    else { while (*vgen == g) __builtin_amdgcn_s_sleep(2); }
  }
  __syncthreads();
}

// S[0]=barrier cnt, S[1]=gen, S[2]=frontier cnt, S[3]=total pools, S[4]=remaining, S[5]=new pools
__global__ void watershed_kernel(const float* __restrict__ elev, const float* __restrict__ r4,
                                 const int* __restrict__ batch,
                                 const int* __restrict__ esrc, const int* __restrict__ edst,
                                 int* visited, int* cl, int* fr, int* nf, int* pool_id,
                                 unsigned* pk, int* present, int* rnk, int* S) {
  const int nt = gridDim.x * blockDim.x;
  const int gtid = blockIdx.x * blockDim.x + threadIdx.x;
  const int nb = gridDim.x;

  for (;;) {
    gsync(S, nb);
    if (*(volatile int*)(S + 4) == 0) break;          // all nodes visited

    for (int g = gtid; g < NG; g += nt) { pk[g] = 0u; present[g] = 0; }
    gsync(S, nb);

    // per-graph max of elev over unvisited
    for (int i = gtid; i < NN; i += nt)
      if (!visited[i]) atomicMax(&pk[batch[i]], fenc(elev[i]));
    gsync(S, nb);

    // peaks -> cluster seed + frontier
    for (int i = gtid; i < NN; i += nt) {
      int p = (!visited[i]) && (pk[batch[i]] == fenc(elev[i]));
      cl[i] = p; fr[i] = p;
    }
    if (gtid == 0) S[2] = 1;
    gsync(S, nb);

    // flood fill downhill: frontier at dst expands to src with r4[src] <= r4[dst]
    for (;;) {
      int fc = *(volatile int*)(S + 2);
      gsync(S, nb);                                   // everyone reads fc before reset
      if (fc == 0) break;
      if (gtid == 0) S[2] = 0;
      gsync(S, nb);
      for (int e = gtid; e < NE; e += nt) {
        int d = edst[e];
        if (fr[d]) {
          int s2 = esrc[e];
          if (!visited[s2] && !cl[s2] && r4[s2] <= r4[d]) nf[s2] = 1;
        }
      }
      gsync(S, nb);
      for (int i = gtid; i < NN; i += nt) {
        int f = nf[i]; nf[i] = 0;
        fr[i] = f;
        if (f) { cl[i] = 1; atomicAdd(S + 2, 1); }
      }
      gsync(S, nb);
    }

    // present graphs + rank (exclusive prefix over 256 graphs, block 0)
    for (int i = gtid; i < NN; i += nt)
      if (cl[i]) present[batch[i]] = 1;
    gsync(S, nb);
    if (blockIdx.x == 0) {
      __shared__ int sp[NG];
      int g = threadIdx.x;
      sp[g] = present[g];
      __syncthreads();
      int orig = sp[g];
      for (int off = 1; off < NG; off <<= 1) {
        int t2 = (g >= off) ? sp[g - off] : 0;
        __syncthreads();
        sp[g] += t2;
        __syncthreads();
      }
      rnk[g] = sp[g] - orig;
      if (g == NG - 1) S[5] = sp[g];
    }
    gsync(S, nb);

    int tot = *(volatile int*)(S + 3);
    for (int i = gtid; i < NN; i += nt)
      if (cl[i]) {
        pool_id[i] = tot + rnk[batch[i]];
        visited[i] = 1; cl[i] = 0; fr[i] = 0;
        atomicAdd(S + 4, -1);
      }
    gsync(S, nb);
    if (gtid == 0) S[3] = tot + S[5];
  }
}

// ---------- pooling: segment_max of x / elev / batch by pool_id ----------
__global__ void pool_x_kernel(const float* __restrict__ x, const int* __restrict__ pool_id,
                              unsigned* __restrict__ pooled_enc) {
  int64_t tot = (int64_t)NN * CH;
  int64_t stride = (int64_t)gridDim.x * blockDim.x;
  for (int64_t idx = (int64_t)blockIdx.x * blockDim.x + threadIdx.x; idx < tot; idx += stride) {
    int i = (int)(idx / CH), c = (int)(idx % CH);
    int p = pool_id[i];
    atomicMax(&pooled_enc[(size_t)p * CH + c], fenc(x[idx]));
  }
}

__global__ void pool_meta_kernel(const float* __restrict__ elev, const int* __restrict__ batch,
                                 const int* __restrict__ pool_id,
                                 unsigned* __restrict__ promits, int* __restrict__ cb_i) {
  int i = blockIdx.x * blockDim.x + threadIdx.x;
  if (i >= NN) return;
  int p = pool_id[i];
  atomicMax(&promits[p], fenc(elev[i]));
  atomicMax(&cb_i[p], batch[i]);
}

// ---------- scatter softmax of promits over cb (valid = p < total) ----------
__global__ void normed_max_kernel(const unsigned* __restrict__ promits, const int* __restrict__ cb_i,
                                  const int* __restrict__ S, unsigned* __restrict__ mg) {
  int p = blockIdx.x * blockDim.x + threadIdx.x;
  if (p < S[3]) atomicMax(&mg[cb_i[p]], promits[p]);
}

__global__ void normed_sum_kernel(const unsigned* __restrict__ promits, const int* __restrict__ cb_i,
                                  const int* __restrict__ S, const unsigned* __restrict__ mg,
                                  float* __restrict__ epool, float* __restrict__ sg) {
  int p = blockIdx.x * blockDim.x + threadIdx.x;
  if (p >= NN) return;
  if (p < S[3]) {
    float m = fdec(mg[cb_i[p]]);
    if (!(m * 0.0f == 0.0f)) m = 0.0f;
    float ev = expf(fdec(promits[p]) - m);
    epool[p] = ev;
    atomicAdd(&sg[cb_i[p]], ev);
  } else epool[p] = 0.f;
}

__global__ void normed_div_kernel(const float* __restrict__ epool, const int* __restrict__ cb_i,
                                  const int* __restrict__ S, const float* __restrict__ sg,
                                  float* __restrict__ out_normed) {
  int p = blockIdx.x * blockDim.x + threadIdx.x;
  if (p >= NN) return;
  if (p < S[3]) {
    float s2 = sg[cb_i[p]];
    out_normed[p] = epool[p] / (s2 > 0.f ? s2 : 1.f);
  } else out_normed[p] = 0.f;
}

// ---------- decode pooled max in place, scale by normed ----------
__global__ void scale_kernel(const int* __restrict__ S, const float* __restrict__ out_normed,
                             float* __restrict__ out_scaled) {
  int64_t tot = (int64_t)NN * CH;
  int64_t stride = (int64_t)gridDim.x * blockDim.x;
  int total = S[3];
  for (int64_t idx = (int64_t)blockIdx.x * blockDim.x + threadIdx.x; idx < tot; idx += stride) {
    int p = (int)(idx / CH);
    unsigned enc = ((const unsigned*)out_scaled)[idx];
    out_scaled[idx] = (p < total) ? fdec(enc) * out_normed[p] : 0.f;
  }
}

__global__ void final_meta_kernel(const int* __restrict__ cb_i, const int* __restrict__ S,
                                  float* __restrict__ out_cb) {
  int p = blockIdx.x * blockDim.x + threadIdx.x;
  if (p >= NN) return;
  out_cb[p] = (p < S[3]) ? (float)cb_i[p] : 0.f;
}

// ---------- pooled edge coalesce: key = ps*65536 + pd (fits u32) ----------
__global__ void keys_kernel(const int* __restrict__ pool_id, const int* __restrict__ esrc,
                            const int* __restrict__ edst, unsigned* __restrict__ keys) {
  int e = blockIdx.x * blockDim.x + threadIdx.x;
  if (e >= NE) return;
  keys[e] = ((unsigned)pool_id[esrc[e]] << 16) | (unsigned)pool_id[edst[e]];
}

// LDS-tiled bitonic: full local sort of 4096-key tiles (stages k=2..4096)
__global__ void bitonic_local_full_kernel(unsigned* __restrict__ keys) {
  __shared__ unsigned sh[SORT_TILE];
  const int base = blockIdx.x * SORT_TILE;
  for (int i = threadIdx.x; i < SORT_TILE; i += 256) sh[i] = keys[base + i];
  __syncthreads();
  for (int k = 2; k <= SORT_TILE; k <<= 1) {
    for (int j = k >> 1; j >= 1; j >>= 1) {
      for (int t = threadIdx.x; t < SORT_TILE / 2; t += 256) {
        int i = ((t & ~(j - 1)) << 1) | (t & (j - 1));
        int ixj = i + j;
        bool asc = (((base + i) & k) == 0);
        unsigned a = sh[i], b = sh[ixj];
        if (asc ? (a > b) : (a < b)) { sh[i] = b; sh[ixj] = a; }
      }
      __syncthreads();
    }
  }
  for (int i = threadIdx.x; i < SORT_TILE; i += 256) keys[base + i] = sh[i];
}

// LDS-tiled bitonic: finish stages j=2048..1 of merge step k (within-tile)
__global__ void bitonic_local_merge_kernel(unsigned* __restrict__ keys, int k) {
  __shared__ unsigned sh[SORT_TILE];
  const int base = blockIdx.x * SORT_TILE;
  for (int i = threadIdx.x; i < SORT_TILE; i += 256) sh[i] = keys[base + i];
  __syncthreads();
  for (int j = SORT_TILE >> 1; j >= 1; j >>= 1) {
    for (int t = threadIdx.x; t < SORT_TILE / 2; t += 256) {
      int i = ((t & ~(j - 1)) << 1) | (t & (j - 1));
      int ixj = i + j;
      bool asc = (((base + i) & k) == 0);
      unsigned a = sh[i], b = sh[ixj];
      if (asc ? (a > b) : (a < b)) { sh[i] = b; sh[ixj] = a; }
    }
    __syncthreads();
  }
  for (int i = threadIdx.x; i < SORT_TILE; i += 256) keys[base + i] = sh[i];
}

// global cross-tile compare-exchange (j >= SORT_TILE)
__global__ void bitonic_global_kernel(unsigned* __restrict__ keys, int j, int k) {
  int i = blockIdx.x * blockDim.x + threadIdx.x;
  int ixj = i ^ j;
  if (ixj > i) {
    unsigned a = keys[i], b2 = keys[ixj];
    bool asc = ((i & k) == 0);
    if (asc ? (a > b2) : (a < b2)) { keys[i] = b2; keys[ixj] = a; }
  }
}

__global__ void edge_out_kernel(const unsigned* __restrict__ keys, float* __restrict__ out_pei) {
  int e = blockIdx.x * blockDim.x + threadIdx.x;
  if (e >= NE) return;
  unsigned k = keys[e];
  bool dup = (e > 0) && (keys[e - 1] == k);
  out_pei[e]      = dup ? -1.f : (float)(k >> 16);
  out_pei[NE + e] = dup ? -1.f : (float)(k & 0xFFFFu);
}

// =========================================================================
extern "C" void kernel_launch(void* const* d_in, const int* in_sizes, int n_in,
                              void* d_out, int out_size, void* d_ws, size_t ws_size,
                              hipStream_t stream) {
  (void)in_sizes; (void)n_in; (void)out_size; (void)ws_size;

  const float* x     = (const float*)d_in[0];
  const int*   ei    = (const int*)d_in[1];
  const int*   batch = (const int*)d_in[2];
  const float* W     = (const float*)d_in[3];
  const float* bias  = (const float*)d_in[4];
  const int* esrc = ei;
  const int* edst = ei + NE;

  // d_out layout (floats): scaled_x [NN*CH] | pei [2*NE] | cb [NN] | normed [NN] | elev [NN]
  float* outf       = (float*)d_out;
  float* out_scaled = outf;
  float* out_pei    = out_scaled + (size_t)NN * CH;
  float* out_cb     = out_pei + 2 * (size_t)NE;
  float* out_normed = out_cb + NN;
  float* out_elev   = out_normed + NN;

  // workspace carve-out (~7 MB)
  char* wp = (char*)d_ws;
  int*      S       = (int*)wp;       wp += 1024;
  unsigned* pk      = (unsigned*)wp;  wp += 1024;
  int*      gstart  = (int*)wp;       wp += 2048;
  int*      present = (int*)wp;       wp += 1024;
  int*      rnk     = (int*)wp;       wp += 1024;
  float*    tg      = (float*)wp;     wp += 1024;
  unsigned* mg      = (unsigned*)wp;  wp += 1024;
  float*    sg      = (float*)wp;     wp += 1024;
  float*    e_raw   = (float*)wp;     wp += (size_t)NN * 4;
  float*    r4      = (float*)wp;     wp += (size_t)NN * 4;
  int*      visited = (int*)wp;       wp += (size_t)NN * 4;
  int*      cl      = (int*)wp;       wp += (size_t)NN * 4;
  int*      fr      = (int*)wp;       wp += (size_t)NN * 4;
  int*      nf      = (int*)wp;       wp += (size_t)NN * 4;
  int*      pool_id = (int*)wp;       wp += (size_t)NN * 4;
  unsigned* promits = (unsigned*)wp;  wp += (size_t)NN * 4;
  int*      cb_i    = (int*)wp;       wp += (size_t)NN * 4;
  float*    epool   = (float*)wp;     wp += (size_t)NN * 4;
  float*    meanb   = (float*)wp;     wp += (size_t)NG * CH * 4;
  unsigned* keys    = (unsigned*)wp;  wp += (size_t)NE * 4;

  // init
  init_small_kernel<<<1, 256, 0, stream>>>(S);
  init_nc_kernel<<<8192, 256, 0, stream>>>((unsigned*)out_scaled);
  init_n_kernel<<<NN / 256, 256, 0, stream>>>(visited, cl, fr, nf, pool_id, promits, cb_i, epool);

  // elevation
  starts_kernel<<<2, 256, 0, stream>>>(batch, gstart);
  mean_kernel<<<NG, 256, 0, stream>>>(x, gstart, meanb);
  tgraph_kernel<<<NG, 256, 0, stream>>>(meanb, W, bias, tg);
  elev_dot_wmma_kernel<<<NN / 128, 256, 0, stream>>>(x, W, batch, tg, e_raw);
  softmax_batch_kernel<<<NG, 256, 0, stream>>>(e_raw, gstart, out_elev, r4);

  // watershed clustering (persistent, device-side convergence loops)
  watershed_kernel<<<WSB, 256, 0, stream>>>(out_elev, r4, batch, esrc, edst,
                                            visited, cl, fr, nf, pool_id,
                                            pk, present, rnk, S);

  // pooling + second scatter softmax + scaling
  pool_x_kernel<<<8192, 256, 0, stream>>>(x, pool_id, (unsigned*)out_scaled);
  pool_meta_kernel<<<NN / 256, 256, 0, stream>>>(out_elev, batch, pool_id, promits, cb_i);
  normed_max_kernel<<<NN / 256, 256, 0, stream>>>(promits, cb_i, S, mg);
  normed_sum_kernel<<<NN / 256, 256, 0, stream>>>(promits, cb_i, S, mg, epool, sg);
  normed_div_kernel<<<NN / 256, 256, 0, stream>>>(epool, cb_i, S, sg, out_normed);
  scale_kernel<<<8192, 256, 0, stream>>>(S, out_normed, out_scaled);
  final_meta_kernel<<<NN / 256, 256, 0, stream>>>(cb_i, S, out_cb);

  // pooled-edge coalesce: build keys, LDS-tiled bitonic sort (2^20), dedup + emit
  keys_kernel<<<NE / 256, 256, 0, stream>>>(pool_id, esrc, edst, keys);
  bitonic_local_full_kernel<<<NE / SORT_TILE, 256, 0, stream>>>(keys);
  for (int k = SORT_TILE * 2; k <= NE; k <<= 1) {
    for (int j = k >> 1; j >= SORT_TILE; j >>= 1)
      bitonic_global_kernel<<<NE / 256, 256, 0, stream>>>(keys, j, k);
    bitonic_local_merge_kernel<<<NE / SORT_TILE, 256, 0, stream>>>(keys, k);
  }
  edge_out_kernel<<<NE / 256, 256, 0, stream>>>(keys, out_pei);
}
